// LinformerAttention_6820408066336
// MI455X (gfx1250) — matrix operand order repaired and checked by reading, not compiled, not run on hardware
//
#include <hip/hip_runtime.h>
#include <hip/hip_bf16.h>

typedef unsigned short u16;
typedef unsigned int   u32;
typedef __attribute__((ext_vector_type(16))) __bf16 bf16x16;
typedef __attribute__((ext_vector_type(8)))  float  f32x8;

union FragAB { bf16x16 v; u32 u[8]; };
union FragC  { f32x8 v; float f[8]; };

__device__ __forceinline__ u16 f32_to_bf16(float f) {
  u32 u = __float_as_uint(f);
  u32 r = u + 0x7fffu + ((u >> 16) & 1u);   // round-to-nearest-even
  return (u16)(r >> 16);
}
__device__ __forceinline__ u32 pack2(u16 lo, u16 hi) { return (u32)lo | ((u32)hi << 16); }

// ---------------------------------------------------------------------------
// Prepass conversions
// ---------------------------------------------------------------------------
__global__ void cvt_f32_bf16_v4(const float4* __restrict__ src, u32* __restrict__ dst, long n4) {
  long i = (long)blockIdx.x * blockDim.x + threadIdx.x;
  if (i < n4) {
    float4 v = src[i];
    dst[2 * i]     = pack2(f32_to_bf16(v.x), f32_to_bf16(v.y));
    dst[2 * i + 1] = pack2(f32_to_bf16(v.z), f32_to_bf16(v.w));
  }
}

// src: rows x cols (f32) -> dst: cols x rows (bf16)
__global__ void cvtT_f32_bf16(const float* __restrict__ src, u16* __restrict__ dst,
                              int rows, int cols) {
  long i = (long)blockIdx.x * blockDim.x + threadIdx.x;
  long n = (long)rows * cols;
  if (i < n) {
    int r = (int)(i / cols), c = (int)(i % cols);
    dst[(long)c * rows + r] = f32_to_bf16(src[i]);
  }
}

// ---------------------------------------------------------------------------
// bf16 WMMA GEMM: C(MxN) = A(MxK) @ B(KxN) [+bias], batched over z.
// 256 threads / 8 waves in a 4x2 grid; block tile 128x128; wave tile 32x64
// (2 M-tiles x 4 N-tiles = 8 WMMAs per K-step from 6 fragments).
// Double-buffered LDS, b128 staging, hoisted address registers.
// ---------------------------------------------------------------------------
__global__ void __launch_bounds__(256)
wmma_gemm_bf16(const u16* __restrict__ A, const u16* __restrict__ B,
               const float* __restrict__ bias,
               u16* __restrict__ Cb, float* __restrict__ Cf,
               int M, int N, int K, int lda, int ldb, int ldc,
               long strideA, long strideB, long strideC) {
  __shared__ u32 Asu[2][128][16];   // A tile: 128 rows x 32 K (bf16 pairs)
  __shared__ u32 BtU[2][128][20];   // B tile transposed: 128 cols x 16 K-pairs (80B rows)

  const int tid  = threadIdx.x;
  const int lane = tid & 31;
  const int wave = tid >> 5;
  const int wm   = wave >> 1;       // 0..3
  const int wn   = wave & 1;        // 0..1
  const int bm   = blockIdx.y * 128;
  const int bn   = blockIdx.x * 128;
  const int mloc = lane & 15;
  const int half = lane >> 4;

  const u32* A32 = (const u32*)(A + (long)blockIdx.z * strideA);
  const u32* B32 = (const u32*)(B + (long)blockIdx.z * strideB);
  const int ldau = lda >> 1, ldbu = ldb >> 1;

  // A staging: 2 units/thread (rows arow, arow+64; 16B each)
  const int arow = tid >> 2;                 // 0..63
  const int aqi  = (tid & 3) * 4;
  long aidx0 = (long)(bm + arow) * ldau + aqi;
  long aidx1 = (long)(bm + arow + 64) * ldau + aqi;
  // B staging: 1 unit/thread (K-pair kp, 8 columns at np4)
  const int kp  = tid >> 4;                  // 0..15
  const int np4 = tid & 15;                  // 0..15
  long bidx0 = (long)(2 * kp) * ldbu + (bn >> 1) + np4 * 4;
  long bidx1 = bidx0 + ldbu;
  const long bstep = (long)16 * ldbu;

  uint4 ra0, ra1, rb0, rb1;
  auto load_tiles = [&]() {
    ra0 = *(const uint4*)(A32 + aidx0);
    ra1 = *(const uint4*)(A32 + aidx1);
    rb0 = *(const uint4*)(B32 + bidx0);
    rb1 = *(const uint4*)(B32 + bidx1);
    aidx0 += 16; aidx1 += 16; bidx0 += bstep; bidx1 += bstep;
  };
  auto store_tiles = [&](int buf) {
    *(uint4*)&Asu[buf][arow][aqi]      = ra0;
    *(uint4*)&Asu[buf][arow + 64][aqi] = ra1;
    u32 c0[4] = {rb0.x, rb0.y, rb0.z, rb0.w};
    u32 c1[4] = {rb1.x, rb1.y, rb1.z, rb1.w};
#pragma unroll
    for (int j = 0; j < 4; ++j) {
      int col = 8 * np4 + 2 * j;
      BtU[buf][col][kp]     = (c0[j] & 0xffffu) | (c1[j] << 16);
      BtU[buf][col + 1][kp] = (c0[j] >> 16)     | (c1[j] & 0xffff0000u);
    }
  };

  FragC acc[2][4];
#pragma unroll
  for (int m = 0; m < 2; ++m)
#pragma unroll
    for (int n = 0; n < 4; ++n)
#pragma unroll
      for (int r = 0; r < 8; ++r) acc[m][n].f[r] = 0.f;

  load_tiles();
  store_tiles(0);
  __syncthreads();

  const int nk = K >> 5;
  for (int it = 0; it < nk; ++it) {
    const int buf = it & 1;
    const bool more = (it + 1) < nk;
    if (more) load_tiles();               // next tile's VMEM in flight over WMMAs

    // Two A fragments (wave rows wm*32 + {0,16})
    FragAB af[2];
#pragma unroll
    for (int m = 0; m < 2; ++m) {
      int ar = wm * 32 + m * 16 + mloc, kb2 = half * 4;
#pragma unroll
      for (int i = 0; i < 4; ++i) {
        af[m].u[i]     = Asu[buf][ar][kb2 + i];
        af[m].u[4 + i] = Asu[buf][ar][8 + kb2 + i];
      }
    }
    // Four B fragments (wave cols wn*64 + n*16), each reused for 2 WMMAs
#pragma unroll
    for (int n = 0; n < 4; ++n) {
      FragAB bfr;
      int ncol = wn * 64 + n * 16 + mloc;
#pragma unroll
      for (int v = 0; v < 8; ++v)
        bfr.u[v] = BtU[buf][ncol][half * 4 + v + (v & 4)];
#pragma unroll
      for (int m = 0; m < 2; ++m)
        acc[m][n].v = __builtin_amdgcn_wmma_f32_16x16x32_bf16(
            false, af[m].v, false, bfr.v, (short)0, acc[m][n].v, false, false);
    }

    if (more) store_tiles(buf ^ 1);
    __syncthreads();
  }

  // Epilogue: pack lane pairs -> 32-bit (bf16) / 64-bit (fp32) stores
  long zC = (long)blockIdx.z * strideC;
#pragma unroll
  for (int n = 0; n < 4; ++n) {
    int col = bn + wn * 64 + n * 16 + mloc;
    float bv = bias ? bias[col] : 0.f;
#pragma unroll
    for (int m = 0; m < 2; ++m) {
#pragma unroll
      for (int r = 0; r < 8; ++r) {
        int row = bm + wm * 32 + m * 16 + r + (half << 3);
        float v = acc[m][n].f[r] + bv;
        long off = zC + (long)row * ldc + col;
        if (Cf) {
          float ov = __shfl_xor(v, 1, 32);
          if ((mloc & 1) == 0) {
            float2 pr; pr.x = v; pr.y = ov;
            *(float2*)(Cf + off) = pr;
          }
        } else {
          u32 myb = f32_to_bf16(v);
          u32 ob  = (u32)__shfl_xor((int)myb, 1, 32);
          if ((mloc & 1) == 0)
            ((u32*)Cb)[off >> 1] = pack2((u16)myb, (u16)ob);
        }
      }
    }
  }
}

// ---------------------------------------------------------------------------
// Fused Linformer attention (streaming softmax), double-buffered k/v chunks.
// ---------------------------------------------------------------------------
__global__ void __launch_bounds__(128)
linformer_attn(const u16* __restrict__ qkv, const u16* __restrict__ kproj,
               const u16* __restrict__ vproj, u16* __restrict__ Y) {
  __shared__ u32 qsu[64][32];          // q tile: 64 t x 64 d
  __shared__ u32 kps[2][32][32];       // kproj chunk: 32 slots x 64 d
  __shared__ u32 vtU[2][64][20];       // vproj chunk transposed: 64 d x 16 slot-pairs
  __shared__ u32 psu[4][16][16];       // per-wave P tile (same-wave use only)

  const int tid  = threadIdx.x;
  const int lane = tid & 31;
  const int wave = tid >> 5;
  const int mloc = lane & 15;
  const int half = lane >> 4;
  const int t0   = blockIdx.x * 64;
  const int h    = blockIdx.y;
  const int b    = blockIdx.z;

  // ---- one-time q tile staging (b128) ----
  {
    const u32* qp = (const u32*)qkv + (((long)(b * 4096 + t0) * 3072 + h * 64) >> 1);
#pragma unroll
    for (int i = 0; i < 4; ++i) {
      int u = tid + i * 128;
      int r = u >> 3, qi = (u & 7) * 4;
      *(uint4*)&qsu[r][qi] = *(const uint4*)(qp + (long)r * 1536 + qi);
    }
  }

  // ---- per-thread k/v staging coordinates (hoisted) ----
  const u32* kpP = (const u32*)kproj + ((long)b * 524288 + h * 32);
  const u32* vpP = (const u32*)vproj + ((long)b * 524288 + h * 32);
  const int rr  = tid >> 3;            // 0..15 (+16)
  const int qi  = (tid & 7) * 4;
  long koff0 = (long)rr * 512 + qi;
  long koff1 = (long)(rr + 16) * 512 + qi;
  const int kp  = tid >> 3;            // slot-pair 0..15
  const int np4 = tid & 7;
  long voff0 = (long)(2 * kp) * 512 + np4 * 4;
  long voff1 = voff0 + 512;

  uint4 rk0, rk1, rv0, rv1;
  auto load_kv = [&]() {
    rk0 = *(const uint4*)(kpP + koff0);
    rk1 = *(const uint4*)(kpP + koff1);
    rv0 = *(const uint4*)(vpP + voff0);
    rv1 = *(const uint4*)(vpP + voff1);
    koff0 += 16384; koff1 += 16384; voff0 += 16384; voff1 += 16384;  // +32 slots
  };
  auto store_kv = [&](int buf) {
    *(uint4*)&kps[buf][rr][qi]      = rk0;
    *(uint4*)&kps[buf][rr + 16][qi] = rk1;
    u32 c0[4] = {rv0.x, rv0.y, rv0.z, rv0.w};
    u32 c1[4] = {rv1.x, rv1.y, rv1.z, rv1.w};
#pragma unroll
    for (int j = 0; j < 4; ++j) {
      int col = 8 * np4 + 2 * j;
      vtU[buf][col][kp]     = (c0[j] & 0xffffu) | (c1[j] << 16);
      vtU[buf][col + 1][kp] = (c0[j] >> 16)     | (c1[j] & 0xffff0000u);
    }
  };

  load_kv();
  store_kv(0);
  __syncthreads();   // q tile + chunk0 visible

  // Persistent q A-fragments (two 32-wide d-chunks)
  FragAB aq[2];
  {
    int trow = wave * 16 + mloc, kb2 = half * 4;
#pragma unroll
    for (int dc = 0; dc < 2; ++dc)
#pragma unroll
      for (int i = 0; i < 4; ++i) {
        aq[dc].u[i]     = qsu[trow][dc * 16 + kb2 + i];
        aq[dc].u[4 + i] = qsu[trow][dc * 16 + 8 + kb2 + i];
      }
  }

  FragC acc[4];
  float mrow[8], lrow[8];
#pragma unroll
  for (int n = 0; n < 4; ++n)
#pragma unroll
    for (int r = 0; r < 8; ++r) acc[n].f[r] = 0.f;
#pragma unroll
  for (int r = 0; r < 8; ++r) { mrow[r] = -1e30f; lrow[r] = 0.f; }

  const float scale = 0.125f;          // 1/sqrt(64)
  int nchunk = ((t0 + 63) >> 5) + 1;   // causal: chunk active iff ch*32 <= t0+63
  if (nchunk > 32) nchunk = 32;

  for (int ch = 0; ch < nchunk; ++ch) {
    const int buf = ch & 1;
    const bool more = (ch + 1) < nchunk;
    if (more) load_kv();               // next chunk VMEM in flight

    // S tiles: preload all 4 B fragments (table k order), then 4 WMMAs
    FragAB bk[2][2];
#pragma unroll
    for (int tile = 0; tile < 2; ++tile) {
      int slot_loc = tile * 16 + mloc;
#pragma unroll
      for (int dc = 0; dc < 2; ++dc)
#pragma unroll
        for (int v = 0; v < 8; ++v)
          bk[tile][dc].u[v] = kps[buf][slot_loc][dc * 16 + half * 4 + v + (v & 4)];
    }
    FragC s[2];
#pragma unroll
    for (int tile = 0; tile < 2; ++tile) {
#pragma unroll
      for (int r = 0; r < 8; ++r) s[tile].f[r] = 0.f;
#pragma unroll
      for (int dc = 0; dc < 2; ++dc)
        s[tile].v = __builtin_amdgcn_wmma_f32_16x16x32_bf16(
            false, aq[dc].v, false, bk[tile][dc].v, (short)0, s[tile].v, false, false);
    }

    // Scale + causal mask + row-max (reduce across 16-lane half)
    const int j0 = ch * 32 + mloc;
    const int j1 = j0 + 16;
    float rmax[8];
#pragma unroll
    for (int r = 0; r < 8; ++r) {
      int trow_g = t0 + wave * 16 + r + (half << 3);
      float a0 = (j0 > trow_g) ? -1e30f : s[0].f[r] * scale;
      float a1 = (j1 > trow_g) ? -1e30f : s[1].f[r] * scale;
      s[0].f[r] = a0; s[1].f[r] = a1;
      rmax[r] = fmaxf(a0, a1);
    }
#pragma unroll
    for (int msk = 1; msk <= 8; msk <<= 1)
#pragma unroll
      for (int r = 0; r < 8; ++r)
        rmax[r] = fmaxf(rmax[r], __shfl_xor(rmax[r], msk, 32));

    // Online softmax; P replaces s[]
    float alpha[8], psum[8];
#pragma unroll
    for (int r = 0; r < 8; ++r) {
      int trow_g = t0 + wave * 16 + r + (half << 3);
      float nm = fmaxf(mrow[r], rmax[r]);
      alpha[r] = __expf(mrow[r] - nm);
      mrow[r]  = nm;
      float p0 = (j0 > trow_g) ? 0.f : __expf(s[0].f[r] - nm);
      float p1 = (j1 > trow_g) ? 0.f : __expf(s[1].f[r] - nm);
      s[0].f[r] = p0; s[1].f[r] = p1;
      psum[r] = p0 + p1;
    }
#pragma unroll
    for (int msk = 1; msk <= 8; msk <<= 1)
#pragma unroll
      for (int r = 0; r < 8; ++r)
        psum[r] += __shfl_xor(psum[r], msk, 32);
#pragma unroll
    for (int r = 0; r < 8; ++r) lrow[r] = lrow[r] * alpha[r] + psum[r];
#pragma unroll
    for (int n = 0; n < 4; ++n)
#pragma unroll
      for (int r = 0; r < 8; ++r) acc[n].f[r] *= alpha[r];

    // P: C-layout -> psu (packed u32, same-wave only: DS in-order per wave, no barrier)
#pragma unroll
    for (int tile = 0; tile < 2; ++tile) {
#pragma unroll
      for (int r = 0; r < 8; ++r) {
        u32 myb = f32_to_bf16(s[tile].f[r]);
        u32 ob  = (u32)__shfl_xor((int)myb, 1, 32);
        int rloc = r + (half << 3);
        if ((mloc & 1) == (tile & 1)) {
          u32 packed = (mloc & 1) ? pack2((u16)ob, (u16)myb) : pack2((u16)myb, (u16)ob);
          psu[wave][rloc][tile * 8 + (mloc >> 1)] = packed;
        }
      }
    }

    FragAB ap;
    {
      int kb2 = half * 4;
#pragma unroll
      for (int i = 0; i < 4; ++i) {
        ap.u[i]     = psu[wave][mloc][kb2 + i];
        ap.u[4 + i] = psu[wave][mloc][8 + kb2 + i];
      }
    }
    // y += P @ vproj: preload 4 B fragments then 4 WMMAs
    FragAB bv4[4];
#pragma unroll
    for (int n = 0; n < 4; ++n) {
      int ncol = n * 16 + mloc;
#pragma unroll
      for (int v = 0; v < 8; ++v)
        bv4[n].u[v] = vtU[buf][ncol][half * 4 + v + (v & 4)];
    }
#pragma unroll
    for (int n = 0; n < 4; ++n)
      acc[n].v = __builtin_amdgcn_wmma_f32_16x16x32_bf16(
          false, ap.v, false, bv4[n].v, (short)0, acc[n].v, false, false);

    if (more) store_kv(buf ^ 1);
    __syncthreads();
  }

  // Finalize: y /= l, packed bf16 stores into Y at (b*T+t, h*64+d)
#pragma unroll
  for (int n = 0; n < 4; ++n) {
    int col = h * 64 + n * 16 + mloc;
#pragma unroll
    for (int r = 0; r < 8; ++r) {
      int trow_g = t0 + wave * 16 + r + (half << 3);
      float y = acc[n].f[r] / lrow[r];
      u32 myb = f32_to_bf16(y);
      u32 ob  = (u32)__shfl_xor((int)myb, 1, 32);
      if ((mloc & 1) == 0) {
        long off = (long)(b * 4096 + trow_g) * 1024 + col;  // col even here
        ((u32*)Y)[off >> 1] = pack2((u16)myb, (u16)ob);
      }
    }
  }
}

// ---------------------------------------------------------------------------
// Launch
// ---------------------------------------------------------------------------
extern "C" void kernel_launch(void* const* d_in, const int* in_sizes, int n_in,
                              void* d_out, int out_size, void* d_ws, size_t ws_size,
                              hipStream_t stream) {
  const float* x      = (const float*)d_in[0];
  const float* W_attn = (const float*)d_in[1];
  const float* b_attn = (const float*)d_in[2];
  const float* W_proj = (const float*)d_in[3];
  const float* b_proj = (const float*)d_in[4];
  const float* E      = (const float*)d_in[5];
  const float* F      = (const float*)d_in[6];
  float* out = (float*)d_out;

  char* ws = (char*)d_ws;
  u16* Xb    = (u16*)ws; ws += (size_t)16384 * 1024 * 2;
  u16* WAb   = (u16*)ws; ws += (size_t)1024 * 3072 * 2;
  u16* WPb   = (u16*)ws; ws += (size_t)1024 * 1024 * 2;
  u16* Et    = (u16*)ws; ws += (size_t)1024 * 4096 * 2;
  u16* Ft    = (u16*)ws; ws += (size_t)1024 * 4096 * 2;
  u16* qkvb  = (u16*)ws; ws += (size_t)16384 * 3072 * 2;
  u16* kprj  = (u16*)ws; ws += (size_t)4 * 1024 * 1024 * 2;
  u16* vprj  = (u16*)ws; ws += (size_t)4 * 1024 * 1024 * 2;
  u16* Yb    = (u16*)ws; ws += (size_t)16384 * 1024 * 2;

  {
    long n4 = 16384L * 1024 / 4;
    cvt_f32_bf16_v4<<<dim3((unsigned)((n4 + 255) / 256)), 256, 0, stream>>>(
        (const float4*)x, (u32*)Xb, n4);
    n4 = 1024L * 3072 / 4;
    cvt_f32_bf16_v4<<<dim3((unsigned)((n4 + 255) / 256)), 256, 0, stream>>>(
        (const float4*)W_attn, (u32*)WAb, n4);
    n4 = 1024L * 1024 / 4;
    cvt_f32_bf16_v4<<<dim3((unsigned)((n4 + 255) / 256)), 256, 0, stream>>>(
        (const float4*)W_proj, (u32*)WPb, n4);
    long n = 4096L * 1024;
    cvtT_f32_bf16<<<dim3((unsigned)((n + 255) / 256)), 256, 0, stream>>>(E, Et, 4096, 1024);
    cvtT_f32_bf16<<<dim3((unsigned)((n + 255) / 256)), 256, 0, stream>>>(F, Ft, 4096, 1024);
  }

  // qkv = Xb @ WAb + b_attn  (16384 x 3072, K=1024) -> bf16
  wmma_gemm_bf16<<<dim3(3072 / 128, 16384 / 128, 1), 256, 0, stream>>>(
      Xb, WAb, b_attn, qkvb, nullptr,
      16384, 3072, 1024, 1024, 3072, 3072, 0, 0, 0);

  // kproj[b] = E^T @ K_b   (1024 x 1024, K=4096)
  wmma_gemm_bf16<<<dim3(1024 / 128, 1024 / 128, 4), 256, 0, stream>>>(
      Et, qkvb + 1024, nullptr, kprj, nullptr,
      1024, 1024, 4096, 4096, 3072, 1024,
      0, (long)4096 * 3072, (long)1024 * 1024);

  // vproj[b] = F^T @ V_b
  wmma_gemm_bf16<<<dim3(1024 / 128, 1024 / 128, 4), 256, 0, stream>>>(
      Ft, qkvb + 2048, nullptr, vprj, nullptr,
      1024, 1024, 4096, 4096, 3072, 1024,
      0, (long)4096 * 3072, (long)1024 * 1024);

  // Fused attention -> Y (bf16)
  linformer_attn<<<dim3(4096 / 64, 16, 4), 128, 0, stream>>>(qkvb, kprj, vprj, Yb);

  // out = Y @ W_proj + b_proj  (fp32)
  wmma_gemm_bf16<<<dim3(1024 / 128, 16384 / 128, 1), 256, 0, stream>>>(
      Yb, WPb, b_proj, nullptr, out,
      16384, 1024, 1024, 1024, 1024, 1024, 0, 0, 0);
}
